// WikiSQLModel_75960791597733
// MI455X (gfx1250) — matrix operand order repaired and checked by reading, not compiled, not run on hardware
//
#include <hip/hip_runtime.h>
#include <hip/hip_bf16.h>
#include <stdint.h>

typedef __bf16 bf16;
typedef __attribute__((ext_vector_type(16))) __bf16 v16bf;
typedef __attribute__((ext_vector_type(8)))  __bf16 v8bf;
typedef __attribute__((ext_vector_type(8)))  float  v8f;
typedef __attribute__((ext_vector_type(4)))  float  v4f;
typedef __attribute__((ext_vector_type(4)))  unsigned int u32x4;
typedef __attribute__((ext_vector_type(8)))  int i32x8;
typedef __attribute__((ext_vector_type(4)))  int i32x4;

#define D_MODEL 768
#define NH 8
#define HD 96
#define SEQ 512
#define BATCH 64
#define MTOT (BATCH * SEQ)   // 32768
#define NSEG 33
#define COMMA 1010

#if defined(__has_builtin)
#if __has_builtin(__builtin_amdgcn_tensor_load_to_lds)
#define HAVE_TDM 1
#else
#define HAVE_TDM 0
#endif
#else
#define HAVE_TDM 0
#endif

static __device__ __forceinline__ bf16 f2bf(float f) {
  union { float f; uint32_t u; } v; v.f = f;
  uint32_t r = v.u + 0x7FFFu + ((v.u >> 16) & 1u);   // round-to-nearest-even
  uint16_t h = (uint16_t)(r >> 16);
  bf16 o; __builtin_memcpy(&o, &h, 2);
  return o;
}

static __device__ __forceinline__ float redmax16(float v) {
#pragma unroll
  for (int m = 1; m < 16; m <<= 1) v = fmaxf(v, __shfl_xor(v, m, 32));
  return v;
}
static __device__ __forceinline__ float redsum16(float v) {
#pragma unroll
  for (int m = 1; m < 16; m <<= 1) v += __shfl_xor(v, m, 32);
  return v;
}

static __device__ __forceinline__ v8f wmma_bf16(v16bf a, v16bf b, v8f c) {
  return __builtin_amdgcn_wmma_f32_16x16x32_bf16(false, a, false, b, (short)0, c,
                                                 false, false);
}

static __device__ __forceinline__ v16bf load_frag2(const bf16* lo, const bf16* hi) {
  union { v16bf v; v8bf h[2]; } u;
  u.h[0] = *(const v8bf*)lo;
  u.h[1] = *(const v8bf*)hi;
  return u.v;
}

#if HAVE_TDM
// Issue one 2D TDM tile load: tile1 rows x tile0 elements (bf16), row stride
// stride0 (elements), landing compactly in LDS at lds_byte_addr.
// D# bitfields per CDNA5 ISA ch.8 (group0: count/lds/global/type,
// group1: data_size=2B, dims, tile dims, dim0 stride; groups 2/3 zero = 2D).
static __device__ __forceinline__ void tdm_load_2d(uint32_t lds_byte_addr,
                                                   const void* gptr,
                                                   uint32_t dim0, uint32_t dim1,
                                                   uint32_t stride0,
                                                   uint32_t tile0,
                                                   uint32_t tile1) {
  uint64_t ga = (uint64_t)(uintptr_t)gptr;
  u32x4 g0;
  g0[0] = 1u;                                            // count=1, user desc
  g0[1] = lds_byte_addr;                                 // lds_addr
  g0[2] = (uint32_t)(ga & 0xFFFFFFFFu);                  // global_addr[31:0]
  g0[3] = (uint32_t)((ga >> 32) & 0x1FFFFFFu) | (2u << 30);  // ga[56:32], type=2
  i32x8 g1;
  g1[0] = (int)(1u << 16);                               // data_size=1 (2 bytes)
  g1[1] = (int)((dim0 & 0xFFFFu) << 16);                 // tensor_dim0[15:0]
  g1[2] = (int)(((dim0 >> 16) & 0xFFFFu) | ((dim1 & 0xFFFFu) << 16));
  g1[3] = (int)(((dim1 >> 16) & 0xFFFFu) | ((tile0 & 0xFFFFu) << 16));
  g1[4] = (int)(tile1 & 0xFFFFu);                        // tile_dim1; tile_dim2=0
  g1[5] = (int)stride0;                                  // tensor_dim0_stride lo
  g1[6] = 0;                                             // stride0 hi / stride1 lo
  g1[7] = 0;
  i32x4 z4 = {0, 0, 0, 0};
#if __clang_major__ >= 23
  i32x8 z8 = {0, 0, 0, 0, 0, 0, 0, 0};
  __builtin_amdgcn_tensor_load_to_lds(g0, g1, z4, z4, z8, 0);
#else
  __builtin_amdgcn_tensor_load_to_lds(g0, g1, z4, z4, 0);
#endif
}
#endif  // HAVE_TDM

// ---------------------------------------------------------------------------
// Y = X @ W^T + bias ; X:[M,K] , W:[N,K] row-major, M=32768, N=K=768.
// AMODE: 0 -> X is f32 (converted to bf16 on the fly), 1 -> X is bf16
// OMODE: 0 -> Y bf16 row-major [M,N]
//        1 -> Y bf16 per-head transposed  [B][NH][HD][SEQ]  (for V)
//        2 -> Y f32 row-major + BatchNorm per-seq-position sum/sumsq atomics
// Tile: 128x128, BK=32, 256 threads = 8 waves (4m x 2n), wave = 32x64 region.
// ---------------------------------------------------------------------------
template <int AMODE, int OMODE>
__global__ __launch_bounds__(256) void gemm_xwt(
    const void* __restrict__ Xv, const float* __restrict__ W,
    const float* __restrict__ bias, void* __restrict__ Yv,
    float* __restrict__ stat_sum, float* __restrict__ stat_sq) {
  const int K = D_MODEL, N = D_MODEL;
  __shared__ __align__(32) bf16 ldsA[128 * 48];
  __shared__ __align__(32) bf16 ldsB[128 * 48];

  const int tid = threadIdx.x;
  const int lane = tid & 31, wid = tid >> 5;
  const int al = lane & 15, ah = lane >> 4;
  const int mt = (blockIdx.x & 255) * 128;  // 256 m-blocks
  const int nt = (blockIdx.x >> 8) * 128;   // 6 n-blocks
  const int wm = wid & 3, wn = wid >> 2;    // 4 x 2 wave grid
  const int m0 = mt + wm * 32;
  const int n0 = nt + wn * 64;

  v8f acc[2][4] = {};

  for (int kb = 0; kb < K; kb += 32) {
    // ---- stage A tile (128 rows x 32 k) ----
    if (AMODE == 0) {
      const float* X = (const float*)Xv;
#pragma unroll
      for (int i = 0; i < 4; ++i) {
        int c = tid + i * 256;                 // float4 chunk id (1024 total)
        int r = c >> 3, c4 = (c & 7) * 4;
        v4f x = *(const v4f*)(X + (size_t)(mt + r) * K + kb + c4);
        bf16* dst = &ldsA[r * 48 + c4];
        dst[0] = f2bf(x.x); dst[1] = f2bf(x.y);
        dst[2] = f2bf(x.z); dst[3] = f2bf(x.w);
      }
      if (kb + 32 < K)
        __builtin_prefetch(X + (size_t)(mt + (tid >> 1)) * K + kb + 32, 0, 1);
    } else {
      const bf16* X = (const bf16*)Xv;
#pragma unroll
      for (int i = 0; i < 2; ++i) {
        int c = tid + i * 256;                 // v8bf chunk id (512 total)
        int r = c >> 2, off = (c & 3) * 8;
        v8bf x = *(const v8bf*)(X + (size_t)(mt + r) * K + kb + off);
        *(v8bf*)&ldsA[r * 48 + off] = x;
      }
      if (kb + 32 < K)
        __builtin_prefetch((const bf16*)Xv + (size_t)(mt + (tid >> 1)) * K + kb + 32, 0, 1);
    }
    // ---- stage W tile (128 n-rows x 32 k), f32 -> bf16 ----
#pragma unroll
    for (int i = 0; i < 4; ++i) {
      int c = tid + i * 256;
      int r = c >> 3, c4 = (c & 7) * 4;
      v4f x = *(const v4f*)(W + (size_t)(nt + r) * K + kb + c4);
      bf16* dst = &ldsB[r * 48 + c4];
      dst[0] = f2bf(x.x); dst[1] = f2bf(x.y);
      dst[2] = f2bf(x.z); dst[3] = f2bf(x.w);
    }
    if (kb + 32 < K)
      __builtin_prefetch(W + (size_t)(nt + (tid >> 1)) * K + kb + 32, 0, 1);
    __syncthreads();

    // ---- WMMA: 2 A-frags x 4 B-frags ----
    v16bf afr[2];
#pragma unroll
    for (int mi = 0; mi < 2; ++mi) {
      const bf16* base = &ldsA[(wm * 32 + mi * 16 + al) * 48 + ah * 8];
      afr[mi] = load_frag2(base, base + 16);
    }
#pragma unroll
    for (int ni = 0; ni < 4; ++ni) {
      const bf16* bb = &ldsB[(wn * 64 + ni * 16 + al) * 48 + ah * 8];
      v16bf bfr = load_frag2(bb, bb + 16);
#pragma unroll
      for (int mi = 0; mi < 2; ++mi) acc[mi][ni] = wmma_bf16(afr[mi], bfr, acc[mi][ni]);
    }
    __syncthreads();
  }

  // ---- epilogue ----
  if (OMODE == 0) {
    bf16* Y = (bf16*)Yv;
#pragma unroll
    for (int mi = 0; mi < 2; ++mi)
#pragma unroll
      for (int ni = 0; ni < 4; ++ni)
#pragma unroll
        for (int r = 0; r < 8; ++r) {
          int row = m0 + mi * 16 + ah * 8 + r;
          int col = n0 + ni * 16 + al;
          Y[(size_t)row * N + col] = f2bf(acc[mi][ni][r] + bias[col]);
        }
  } else if (OMODE == 1) {
    bf16* Y = (bf16*)Yv;  // [B][NH][HD][SEQ]
#pragma unroll
    for (int mi = 0; mi < 2; ++mi)
#pragma unroll
      for (int ni = 0; ni < 4; ++ni)
#pragma unroll
        for (int r = 0; r < 8; ++r) {
          int row = m0 + mi * 16 + ah * 8 + r;
          int col = n0 + ni * 16 + al;
          int b = row >> 9, s = row & 511;
          int hh = col / HD, dl = col - hh * HD;
          Y[((size_t)(b * NH + hh) * HD + dl) * SEQ + s] =
              f2bf(acc[mi][ni][r] + bias[col]);
        }
  } else {
    float* Y = (float*)Yv;
    float rs[2][8] = {}, rq[2][8] = {};
#pragma unroll
    for (int mi = 0; mi < 2; ++mi)
#pragma unroll
      for (int ni = 0; ni < 4; ++ni)
#pragma unroll
        for (int r = 0; r < 8; ++r) {
          int row = m0 + mi * 16 + ah * 8 + r;
          int col = n0 + ni * 16 + al;
          float v = acc[mi][ni][r] + bias[col];
          Y[(size_t)row * N + col] = v;
          rs[mi][r] += v;
          rq[mi][r] += v * v;
        }
#pragma unroll
    for (int mi = 0; mi < 2; ++mi)
#pragma unroll
      for (int r = 0; r < 8; ++r) {
        float s1 = redsum16(rs[mi][r]);
        float s2 = redsum16(rq[mi][r]);
        if ((lane & 15) == (mi * 8 + r)) {
          int row = m0 + mi * 16 + ah * 8 + r;
          int s = row & 511;
          atomicAdd(&stat_sum[s], s1);
          atomicAdd(&stat_sq[s], s2);
        }
      }
  }
}

// ---------------------------------------------------------------------------
// Flash-style cross attention, one (b,h,128-query chunk) per block.
// 8 waves x 16 queries; K-chunks of 64 keys staged in LDS via the Tensor Data
// Mover (wave 0 issues descriptors, s_wait_tensorcnt, block barrier publishes);
// online softmax; both matmuls on v_wmma_f32_16x16x32_bf16.
// ---------------------------------------------------------------------------
__global__ __launch_bounds__(256) void attn_kernel(
    const bf16* __restrict__ Q,   // [M, D]
    const bf16* __restrict__ Km,  // [M, D]
    const bf16* __restrict__ VT,  // [B][NH][HD][SEQ]
    bf16* __restrict__ O) {       // [M, D]
  __shared__ __align__(32) bf16 ldsK[64 * HD];      // 64 keys x 96
  __shared__ __align__(32) bf16 ldsVT[HD * 64];     // 96 dims x 64 keys
  __shared__ __align__(32) bf16 ldsP[8 * 16 * 64];  // per-wave P tiles

  const int tid = threadIdx.x, lane = tid & 31, wid = tid >> 5;
  const int al = lane & 15, ah = lane >> 4;
  const int bh = blockIdx.x >> 2;  // b*8 + h
  const int qc = blockIdx.x & 3;
  const int b = bh >> 3, h = bh & 7;
  const int q0 = qc * 128 + wid * 16;  // wave's first query position
  const size_t rowbase = (size_t)b * SEQ;
  const bf16* vbase = VT + (size_t)(b * NH + h) * HD * SEQ;
  const float scale = 0.10206207261596575f;  // 1/sqrt(96)

  // resident q fragments: 16 rows x 96 dims = 3 A-frags
  v16bf qf[3];
#pragma unroll
  for (int ks = 0; ks < 3; ++ks) {
    const bf16* base =
        Q + (rowbase + q0 + al) * D_MODEL + h * HD + ks * 32 + ah * 8;
    qf[ks] = load_frag2(base, base + 16);
  }

  v8f oacc[6] = {};
  float mI[8], lI[8];
#pragma unroll
  for (int r = 0; r < 8; ++r) { mI[r] = -1e30f; lI[r] = 0.0f; }

  for (int j = 0; j < SEQ; j += 64) {
    __syncthreads();  // previous chunk fully consumed before overwrite
#if HAVE_TDM
    if (wid == 0) {
      // K chunk: 64 rows x 96 bf16, row stride D_MODEL -> compact in ldsK
      tdm_load_2d((uint32_t)(uintptr_t)&ldsK[0],
                  Km + (rowbase + j) * D_MODEL + h * HD,
                  /*dim0*/ HD, /*dim1*/ 64, /*stride0*/ D_MODEL,
                  /*tile0*/ HD, /*tile1*/ 64);
      // V^T chunk: 96 rows x 64 bf16, row stride SEQ -> compact in ldsVT
      tdm_load_2d((uint32_t)(uintptr_t)&ldsVT[0], vbase + j,
                  /*dim0*/ 64, /*dim1*/ HD, /*stride0*/ SEQ,
                  /*tile0*/ 64, /*tile1*/ HD);
      __builtin_amdgcn_s_wait_tensorcnt(0);
    }
#else
    // cooperative staging fallback
#pragma unroll
    for (int i = 0; i < 3; ++i) {
      int c = tid + i * 256;  // 768 v8bf chunks
      int r = c / 12, off = (c % 12) * 8;
      v8bf x = *(const v8bf*)(Km + (rowbase + j + r) * D_MODEL + h * HD + off);
      *(v8bf*)&ldsK[r * HD + off] = x;
    }
#pragma unroll
    for (int i = 0; i < 3; ++i) {
      int c = tid + i * 256;
      int d = c >> 3, off = (c & 7) * 8;
      v8bf x = *(const v8bf*)(vbase + (size_t)d * SEQ + j + off);
      *(v8bf*)&ldsVT[d * 64 + off] = x;
    }
#endif
    __syncthreads();

    // scores: 16 queries x 64 keys, K = 96 (3 wmma steps)
    v8f sacc[4] = {};
#pragma unroll
    for (int ks = 0; ks < 3; ++ks) {
#pragma unroll
      for (int nt2 = 0; nt2 < 4; ++nt2) {
        const bf16* kbp = &ldsK[(nt2 * 16 + al) * HD + ks * 32 + ah * 8];
        v16bf bfr = load_frag2(kbp, kbp + 16);
        sacc[nt2] = wmma_bf16(qf[ks], bfr, sacc[nt2]);
      }
    }

    // online softmax update
    float mNew[8], corr[8];
#pragma unroll
    for (int r = 0; r < 8; ++r) {
      float mx = -1e30f;
#pragma unroll
      for (int nt2 = 0; nt2 < 4; ++nt2) mx = fmaxf(mx, sacc[nt2][r] * scale);
      mx = redmax16(mx);
      mNew[r] = fmaxf(mI[r], mx);
      corr[r] = __expf(mI[r] - mNew[r]);
      mI[r] = mNew[r];
    }
    bf16* pbase = &ldsP[wid * 16 * 64];
#pragma unroll
    for (int r = 0; r < 8; ++r) {
      float rsum = 0.0f;
#pragma unroll
      for (int nt2 = 0; nt2 < 4; ++nt2) {
        float p = __expf(sacc[nt2][r] * scale - mNew[r]);
        rsum += p;
        pbase[(ah * 8 + r) * 64 + nt2 * 16 + al] = f2bf(p);
      }
      rsum = redsum16(rsum);
      lI[r] = lI[r] * corr[r] + rsum;
#pragma unroll
      for (int dt = 0; dt < 6; ++dt) oacc[dt][r] *= corr[r];
    }
    __syncthreads();

    // O += P @ V  (K = 64 keys -> 2 wmma steps, 6 d-tiles)
#pragma unroll
    for (int ks2 = 0; ks2 < 2; ++ks2) {
      const bf16* pb = pbase + al * 64 + ks2 * 32 + ah * 8;
      v16bf pf = load_frag2(pb, pb + 16);
#pragma unroll
      for (int dt = 0; dt < 6; ++dt) {
        const bf16* vbp = &ldsVT[(dt * 16 + al) * 64 + ks2 * 32 + ah * 8];
        v16bf vf = load_frag2(vbp, vbp + 16);
        oacc[dt] = wmma_bf16(pf, vf, oacc[dt]);
      }
    }
  }

  // finalize: divide by l, store bf16 row-major
#pragma unroll
  for (int dt = 0; dt < 6; ++dt)
#pragma unroll
    for (int r = 0; r < 8; ++r) {
      float v = oacc[dt][r] / lI[r];
      int row = q0 + ah * 8 + r;
      int col = h * HD + dt * 16 + al;
      O[(rowbase + row) * D_MODEL + col] = f2bf(v);
    }
}

// ---------------------------------------------------------------------------
__global__ void bn_finalize(const float* __restrict__ stat_sum,
                            const float* __restrict__ stat_sq,
                            const float* __restrict__ gamma,
                            const float* __restrict__ beta,
                            float* __restrict__ ig, float* __restrict__ sh) {
  int s = blockIdx.x * blockDim.x + threadIdx.x;
  if (s < SEQ) {
    const float inv_n = 1.0f / (float)(BATCH * D_MODEL);
    float mean = stat_sum[s] * inv_n;
    float var = stat_sq[s] * inv_n - mean * mean;
    float g = gamma[s] * rsqrtf(var + 1e-5f);
    ig[s] = g;
    sh[s] = beta[s] - mean * g;
  }
}

// ret[b,s] = b_out + sum_d (cols + attn_out*ig[s] + sh[s]) * w_out[d]
__global__ __launch_bounds__(256) void head_kernel(
    const float* __restrict__ cols, const float* __restrict__ attn_out,
    const float* __restrict__ ig, const float* __restrict__ sh,
    const float* __restrict__ w_out, const float* __restrict__ b_out,
    float* __restrict__ ret) {
  int lane = threadIdx.x & 31, wid = threadIdx.x >> 5;
  int row = blockIdx.x * 8 + wid;
  int s = row & 511;
  float g = ig[s], c0 = sh[s];
  size_t base = (size_t)row * D_MODEL;
  float acc = 0.0f;
#pragma unroll 4
  for (int it = 0; it < D_MODEL / 32; ++it) {
    int d = it * 32 + lane;
    acc += (cols[base + d] + attn_out[base + d] * g + c0) * w_out[d];
  }
#pragma unroll
  for (int m = 1; m < 32; m <<= 1) acc += __shfl_xor(acc, m, 32);
  if (lane == 0) ret[row] = acc + b_out[0];
}

// comma-delimited segment sum: one wave per batch row
__global__ void segsum_kernel(const int* __restrict__ ids,
                              const float* __restrict__ ret,
                              float* __restrict__ out) {
  int b = blockIdx.x, lane = threadIdx.x;
  int carried = 0;
  for (int c = 0; c < SEQ / 32; ++c) {
    int s = c * 32 + lane;
    int id = ids[b * SEQ + s];
    bool comma = (id == COMMA);
    unsigned mask32 = (unsigned)__ballot(comma);
    int below = __popc(mask32 & ((1u << lane) - 1u));
    if (!comma) {
      atomicAdd(&out[b * NSEG + carried + below], ret[b * SEQ + s]);
    }
    carried += __popc(mask32);
  }
}

// ---------------------------------------------------------------------------
extern "C" void kernel_launch(void* const* d_in, const int* in_sizes, int n_in,
                              void* d_out, int out_size, void* d_ws,
                              size_t ws_size, hipStream_t stream) {
  const float* text_hs = (const float*)d_in[0];
  const float* cols_hs = (const float*)d_in[1];
  const int* col_ids = (const int*)d_in[2];
  // d_in[3] = n_segments (fixed at 33 for this problem)
  const float* Wq = (const float*)d_in[4];
  const float* bq = (const float*)d_in[5];
  const float* Wk = (const float*)d_in[6];
  const float* bk = (const float*)d_in[7];
  const float* Wv = (const float*)d_in[8];
  const float* bv = (const float*)d_in[9];
  const float* Wo = (const float*)d_in[10];
  const float* bo = (const float*)d_in[11];
  const float* gamma = (const float*)d_in[12];
  const float* beta = (const float*)d_in[13];
  const float* w_out = (const float*)d_in[14];
  const float* b_out = (const float*)d_in[15];

  char* ws = (char*)d_ws;
  size_t off = 0;
  auto alloc = [&](size_t bytes) -> void* {
    void* p = ws + off;
    off = (off + bytes + 255) & ~(size_t)255;
    return p;
  };
  bf16* qb = (bf16*)alloc((size_t)MTOT * D_MODEL * 2);
  bf16* kb = (bf16*)alloc((size_t)MTOT * D_MODEL * 2);
  bf16* vT = (bf16*)alloc((size_t)MTOT * D_MODEL * 2);
  bf16* ob = (bf16*)alloc((size_t)MTOT * D_MODEL * 2);
  float* ao = (float*)alloc((size_t)MTOT * D_MODEL * 4);
  float* stat_sum = (float*)alloc(SEQ * 4);
  float* stat_sq = (float*)alloc(SEQ * 4);
  float* ig = (float*)alloc(SEQ * 4);
  float* sh = (float*)alloc(SEQ * 4);
  float* ret = (float*)alloc((size_t)MTOT * 4);

  hipMemsetAsync(stat_sum, 0, SEQ * 4, stream);
  hipMemsetAsync(stat_sq, 0, SEQ * 4, stream);
  hipMemsetAsync(d_out, 0, (size_t)out_size * sizeof(float), stream);

  dim3 blk(256);
  dim3 gproj(256 * 6);  // (M/128) x (N/128)
  gemm_xwt<0, 0><<<gproj, blk, 0, stream>>>(cols_hs, Wq, bq, qb, nullptr, nullptr);
  gemm_xwt<0, 0><<<gproj, blk, 0, stream>>>(text_hs, Wk, bk, kb, nullptr, nullptr);
  gemm_xwt<0, 1><<<gproj, blk, 0, stream>>>(text_hs, Wv, bv, vT, nullptr, nullptr);

  attn_kernel<<<dim3(BATCH * NH * 4), blk, 0, stream>>>(qb, kb, vT, ob);

  gemm_xwt<1, 2><<<gproj, blk, 0, stream>>>(ob, Wo, bo, ao, stat_sum, stat_sq);

  bn_finalize<<<dim3(2), dim3(256), 0, stream>>>(stat_sum, stat_sq, gamma, beta, ig, sh);
  head_kernel<<<dim3(MTOT / 8), blk, 0, stream>>>(cols_hs, ao, ig, sh, w_out, b_out, ret);
  segsum_kernel<<<dim3(BATCH), dim3(32), 0, stream>>>(col_ids, ret, (float*)d_out);
}
